// GAT_56143812493497
// MI455X (gfx1250) — compile-verified, matrix operand order checked by reading, not run on hardware
//
#include <hip/hip_runtime.h>
#include <cstdint>
#include <cstddef>

// Problem constants (match reference)
#define NN 50000        // nodes
#define NE 800000       // edges
#define NH 4            // heads
#define ND 64           // per-head dim
#define NC 256          // channels = NH*ND = fan-in for every layer
#define MTILES (NN/16)  // 3125, exact
#define NEG_SLOPE 0.2f

// LDS B-tile row stride in ushorts: 256 + 8 pad -> 528B = 132 dwords/row.
// 132 mod 64 = 4, so 16 consecutive rows hit banks 0,4,8,...,60: conflict-free.
#define BSTRIDE (NC + 8)

typedef __attribute__((ext_vector_type(16))) __bf16 v16bf;
typedef __attribute__((ext_vector_type(8)))  float  v8f;

struct FragBits { uint4 lo, hi; };

// Load one 16-element bf16 WMMA fragment slice for this lane.
// A (16x32, 16-bit) lane layout: lane = {half = lane>>4, mr = lane&15};
// elements 0..7  hold K = base + half*8 + 0..7
// elements 8..15 hold K = base + 16 + half*8 + 0..7
// 'p' must already point at (row_base + half*8); we add +16 for the hi half.
__device__ __forceinline__ v16bf load_frag16(const unsigned short* p) {
  FragBits f;
  f.lo = *(const uint4*)(p);
  f.hi = *(const uint4*)(p + 16);
  return __builtin_bit_cast(v16bf, f);
}

__device__ __forceinline__ unsigned short f2bf(float x) {
  unsigned int u = __float_as_uint(x);
  u += 0x7FFFu + ((u >> 16) & 1u);   // round-to-nearest-even
  return (unsigned short)(u >> 16);
}

// Sign-split float atomic max -> global_atomic_max_i32 / global_atomic_min_u32
__device__ __forceinline__ void atomicMaxF(float* addr, float val) {
  if (val >= 0.0f) atomicMax((int*)addr, __float_as_int(val));
  else             atomicMin((unsigned int*)addr, __float_as_uint(val));
}

// ---------------- elementwise helpers ----------------

__global__ void gat_fill_f32(float* __restrict__ p, float v, int n) {
  int i = blockIdx.x * blockDim.x + threadIdx.x;
  if (i < n) p[i] = v;
}

__global__ void gat_conv_bf16(unsigned short* __restrict__ dst,
                              const float* __restrict__ src, int n) {
  int i = blockIdx.x * blockDim.x + threadIdx.x;
  if (i < n) dst[i] = f2bf(src[i]);
}

// W [K=256][N=256] row-major f32  ->  Wt [N=256][K=256] bf16 (transposed)
__global__ void gat_conv_w(unsigned short* __restrict__ wt,
                           const float* __restrict__ W) {
  int i = blockIdx.x * blockDim.x + threadIdx.x;  // 65536 threads
  int n = i >> 8, k = i & 255;
  wt[n * NC + k] = f2bf(W[k * NC + n]);
}

// ---------------- bf16 WMMA GEMM: feat[N,256] = hb[N,256] @ Wt^T ----------------
// Block = 128 threads = 4 waves. Wave w -> M-tile (blockIdx.x*4+w),
// N-strip = blockIdx.y*64 .. +63 (exactly one attention head's 64 cols).
// The 64x256 bf16 W strip (32KB) is staged in LDS once per block and shared by
// all 4 waves; each wave preloads its full 16x256 A strip into registers, so
// the 32-WMMA inner loop runs with no global-memory waits.
__global__ void __launch_bounds__(128)
gat_gemm_bf16(const unsigned short* __restrict__ hb,
              const unsigned short* __restrict__ wt,
              float* __restrict__ feat) {
  __shared__ __align__(16) unsigned short Bs[64 * BSTRIDE];  // 33 KB

  const int tid  = threadIdx.x;
  const int wave = tid >> 5;
  const int lane = tid & 31;
  const int n0   = blockIdx.y << 6;

  // Cooperative load of the 64x256 B tile (Wt rows n0..n0+63) into LDS.
  // 2048 uint4 total -> 16 per thread.
#pragma unroll
  for (int it = 0; it < 16; ++it) {
    int v   = it * 128 + tid;      // uint4 index: row*32 + c8
    int row = v >> 5;              // 32 uint4 per 256-ushort row
    int c8  = v & 31;
    uint4 d = *(const uint4*)(wt + (size_t)(n0 + row) * NC + c8 * 8);
    *(uint4*)(&Bs[row * BSTRIDE + c8 * 8]) = d;
  }
  __syncthreads();

  const int mtile = blockIdx.x * 4 + wave;
  if (mtile >= MTILES) return;     // whole-wave exit; EXEC stays all-1s
  const int m0   = mtile << 4;
  const int half = lane >> 4;
  const int mr   = lane & 15;

  // Preload this wave's entire A strip (16 rows x K=256): 8 fragments/lane.
  const unsigned short* arow = hb + (size_t)(m0 + mr) * NC + half * 8;
  v16bf a[8];
#pragma unroll
  for (int t = 0; t < 8; ++t) a[t] = load_frag16(arow + t * 32);

  const unsigned short* b0 = &Bs[( 0 + mr) * BSTRIDE + half * 8];
  const unsigned short* b1 = &Bs[(16 + mr) * BSTRIDE + half * 8];
  const unsigned short* b2 = &Bs[(32 + mr) * BSTRIDE + half * 8];
  const unsigned short* b3 = &Bs[(48 + mr) * BSTRIDE + half * 8];

  v8f acc0 = {}, acc1 = {}, acc2 = {}, acc3 = {};
#pragma unroll
  for (int t = 0; t < 8; ++t) {
    const int kk = t * 32;
    v16bf f0 = load_frag16(b0 + kk);
    v16bf f1 = load_frag16(b1 + kk);
    v16bf f2 = load_frag16(b2 + kk);
    v16bf f3 = load_frag16(b3 + kk);
    acc0 = __builtin_amdgcn_wmma_f32_16x16x32_bf16(false, a[t], false, f0, (short)0, acc0, false, false);
    acc1 = __builtin_amdgcn_wmma_f32_16x16x32_bf16(false, a[t], false, f1, (short)0, acc1, false, false);
    acc2 = __builtin_amdgcn_wmma_f32_16x16x32_bf16(false, a[t], false, f2, (short)0, acc2, false, false);
    acc3 = __builtin_amdgcn_wmma_f32_16x16x32_bf16(false, a[t], false, f3, (short)0, acc3, false, false);
  }

  // C/D layout: lanes 0-15 hold M = m0+0..7 (VGPR r -> M=r), lanes 16-31 hold M = m0+8..15
  const int mbase = m0 + half * 8;
#pragma unroll
  for (int r = 0; r < 8; ++r) {
    float* orow = feat + (size_t)(mbase + r) * NC + mr;
    orow[n0 +  0] = acc0[r];
    orow[n0 + 16] = acc1[r];
    orow[n0 + 32] = acc2[r];
    orow[n0 + 48] = acc3[r];
  }
}

// ---------------- attention pieces ----------------

// el[n,h] = dot(feat[n, h*64 .. h*64+63], al[h]); same for er.
__global__ void gat_el_er(const float* __restrict__ feat,
                          const float* __restrict__ al,
                          const float* __restrict__ ar,
                          float* __restrict__ el, float* __restrict__ er) {
  int i = blockIdx.x * blockDim.x + threadIdx.x;
  if (i >= NN * NH) return;
  int n = i >> 2, h = i & 3;
  const float* fr = feat + (size_t)n * NC + h * ND;
  const float* a0 = al + h * ND;
  const float* a1 = ar + h * ND;
  float sl = 0.f, sr = 0.f;
#pragma unroll 4
  for (int d = 0; d < ND; ++d) {
    float v = fr[d];
    sl += v * a0[d];
    sr += v * a1[d];
  }
  el[i] = sl;
  er[i] = sr;
}

// pass 1: e = leaky_relu(el[src]+er[dst]); per-dst running max (for stable softmax)
__global__ void gat_edge_score(const int* __restrict__ srcv, const int* __restrict__ dstv,
                               const float* __restrict__ el, const float* __restrict__ er,
                               float* __restrict__ exb, float* __restrict__ mb) {
  int i = blockIdx.x * blockDim.x + threadIdx.x;
  if (i >= NE * NH) return;
  int ed = i >> 2, h = i & 3;
  int s = srcv[ed], d = dstv[ed];
  float v = el[s * NH + h] + er[d * NH + h];
  v = (v > 0.f) ? v : NEG_SLOPE * v;
  exb[i] = v;
  atomicMaxF(&mb[d * NH + h], v);
}

// pass 2: ex = exp(e - m[dst]); per-dst sum
__global__ void gat_edge_exp(const int* __restrict__ dstv,
                             float* __restrict__ exb,
                             const float* __restrict__ mb, float* __restrict__ sb) {
  int i = blockIdx.x * blockDim.x + threadIdx.x;
  if (i >= NE * NH) return;
  int ed = i >> 2, h = i & 3;
  int d = dstv[ed];
  float p = __expf(exb[i] - mb[d * NH + h]);
  exb[i] = p;
  atomicAdd(&sb[d * NH + h], p);
}

// pass 3: out[dst] += feat[src] * alpha. One thread per (edge, float4-chunk).
// feat (51MB) + out (51MB) both fit in the 192MB L2 -> L2-resident RMW.
__global__ void gat_edge_aggr(const int* __restrict__ srcv, const int* __restrict__ dstv,
                              const float* __restrict__ exb, const float* __restrict__ sb,
                              const float* __restrict__ feat, float* __restrict__ out) {
  int i = blockIdx.x * blockDim.x + threadIdx.x;
  int ed = i >> 6;           // edge
  int c4 = i & 63;           // which float4 chunk of the 256-channel row
  if (ed >= NE) return;
  int h = c4 >> 4;           // 16 chunks per head
  int s = srcv[ed], d = dstv[ed];
  float denom = fmaxf(sb[d * NH + h], 1e-9f);
  float alpha = exb[ed * NH + h] / denom;
  const float4 f = *(const float4*)(feat + (size_t)s * NC + c4 * 4);
  float* o = out + (size_t)d * NC + c4 * 4;
  atomicAdd(o + 0, f.x * alpha);
  atomicAdd(o + 1, f.y * alpha);
  atomicAdd(o + 2, f.z * alpha);
  atomicAdd(o + 3, f.w * alpha);
}

// epilogue: dst = (act ? elu : id)(out + bias)
__global__ void gat_bias_act(const float* __restrict__ out, const float* __restrict__ b,
                             float* __restrict__ dst, int act) {
  int i = blockIdx.x * blockDim.x + threadIdx.x;
  if (i >= NN * NC) return;
  float x = out[i] + b[i & (NC - 1)];
  if (act) x = (x > 0.f) ? x : (__expf(x) - 1.0f);
  dst[i] = x;
}

// ---------------- host orchestration ----------------

struct Scratch {
  float* feat;           // [NN, NC] f32
  unsigned short* hb;    // [NN, NC] bf16
  unsigned short* wt;    // [NC, NC] bf16 (transposed W)
  float *el, *er, *mb, *sb;  // [NN, NH]
  float* exb;            // [NE, NH]
};

static void run_layer(const float* hin, float* out, float* act_dst,
                      const float* Wl, const float* alh, const float* arh, const float* bh,
                      const int* srcv, const int* dstv, int act,
                      const Scratch& ws, hipStream_t stream) {
  const int TB = 256;
  // weight transpose+convert, activation convert (hin is dead after this + gemm)
  gat_conv_w   <<<(NC * NC + TB - 1) / TB, TB, 0, stream>>>(ws.wt, Wl);
  gat_conv_bf16<<<(NN * NC + TB - 1) / TB, TB, 0, stream>>>(ws.hb, hin, NN * NC);
  // init accumulators (out may alias hin; hb already holds the input)
  gat_fill_f32 <<<(NN * NC + TB - 1) / TB, TB, 0, stream>>>(out, 0.0f, NN * NC);
  gat_fill_f32 <<<(NN * NH + TB - 1) / TB, TB, 0, stream>>>(ws.mb, -3.402823e38f, NN * NH);
  gat_fill_f32 <<<(NN * NH + TB - 1) / TB, TB, 0, stream>>>(ws.sb, 0.0f, NN * NH);
  // GEMM on the matrix cores
  dim3 ggrid((MTILES + 3) / 4, NC / 64);
  gat_gemm_bf16<<<ggrid, 128, 0, stream>>>(ws.hb, ws.wt, ws.feat);
  // attention
  gat_el_er     <<<(NN * NH + TB - 1) / TB, TB, 0, stream>>>(ws.feat, alh, arh, ws.el, ws.er);
  gat_edge_score<<<(NE * NH + TB - 1) / TB, TB, 0, stream>>>(srcv, dstv, ws.el, ws.er, ws.exb, ws.mb);
  gat_edge_exp  <<<(NE * NH + TB - 1) / TB, TB, 0, stream>>>(dstv, ws.exb, ws.mb, ws.sb);
  gat_edge_aggr <<<(NE * 64 + TB - 1) / TB, TB, 0, stream>>>(srcv, dstv, ws.exb, ws.sb, ws.feat, out);
  gat_bias_act  <<<(NN * NC + TB - 1) / TB, TB, 0, stream>>>(out, bh, act_dst, act);
}

extern "C" void kernel_launch(void* const* d_in, const int* in_sizes, int n_in,
                              void* d_out, int out_size, void* d_ws, size_t ws_size,
                              hipStream_t stream) {
  (void)in_sizes; (void)n_in; (void)out_size; (void)ws_size;
  const float* feats = (const float*)d_in[0];
  const int*   srcv  = (const int*)d_in[1];
  const int*   dstv  = (const int*)d_in[2];
  const float* W[3]  = {(const float*)d_in[3], (const float*)d_in[7],  (const float*)d_in[11]};
  const float* al[3] = {(const float*)d_in[4], (const float*)d_in[8],  (const float*)d_in[12]};
  const float* ar[3] = {(const float*)d_in[5], (const float*)d_in[9],  (const float*)d_in[13]};
  const float* bb[3] = {(const float*)d_in[6], (const float*)d_in[10], (const float*)d_in[14]};
  float* outp = (float*)d_out;

  // carve workspace (256B-aligned chunks)
  unsigned char* base = (unsigned char*)d_ws;
  size_t off = 0;
  auto take = [&](size_t bytes) -> void* {
    void* p = base + off;
    off += (bytes + 255) & ~(size_t)255;
    return p;
  };
  Scratch ws;
  ws.feat = (float*)take((size_t)NN * NC * 4);
  ws.hb   = (unsigned short*)take((size_t)NN * NC * 2);
  ws.wt   = (unsigned short*)take((size_t)NC * NC * 2);
  ws.el   = (float*)take((size_t)NN * NH * 4);
  ws.er   = (float*)take((size_t)NN * NH * 4);
  ws.mb   = (float*)take((size_t)NN * NH * 4);
  ws.sb   = (float*)take((size_t)NN * NH * 4);
  ws.exb  = (float*)take((size_t)NE * NH * 4);
  float* hA = (float*)take((size_t)NN * NC * 4);  // ping-pong activation buffer

  // Layer 1: in = features (read-only input), out/act -> hA, ELU
  run_layer(feats, hA, hA, W[0], al[0], ar[0], bb[0], srcv, dstv, 1, ws, stream);
  // Layer 2: in = hA; hA is dead after the bf16 convert, so reuse it as the
  // accumulator. out/act -> hA, ELU
  run_layer(hA, hA, hA, W[1], al[1], ar[1], bb[1], srcv, dstv, 1, ws, stream);
  // Layer 3: in = hA, accumulate into d_out, bias only (no activation)
  run_layer(hA, outp, outp, W[2], al[2], ar[2], bb[2], srcv, dstv, 0, ws, stream);
}
